// TransformerBlock_81003083203503
// MI455X (gfx1250) — compile-verified, matrix operand order checked by reading, not run on hardware
//
#include <hip/hip_runtime.h>
#include <hip/hip_bf16.h>
#include <cmath>

// ---------------------------------------------------------------------------
// CDNA5 (gfx1250) transformer decoder block.
// All matmuls use V_WMMA_F32_16X16X32_BF16 (wave32), fp32 accumulate.
// Shapes: B=2, S=SC=2048, E=1024, H=16, D=64, NT=B*S=4096.
// ---------------------------------------------------------------------------

typedef __attribute__((ext_vector_type(16))) __bf16 v16bf;
typedef __attribute__((ext_vector_type(8)))  float  v8f;

union FragBf {
    v16bf v;
    unsigned int u[8];
};

// K-index for dword p (pair of bf16) of a 16-bit WMMA operand fragment.
// ISA 7.12.2: lanes 0-15 hold K {0..7,16..23}; lanes 16-31 hold K {8..15,24..31}.
__device__ __forceinline__ int kmap(int p, int hi) {
    return 2 * p + ((p >= 4) ? 8 : 0) + hi * 8;
}

__device__ __forceinline__ v8f wmma_bf16(const FragBf& a, const FragBf& b, v8f c) {
    return __builtin_amdgcn_wmma_f32_16x16x32_bf16(
        false, a.v, false, b.v, (short)0, c, false, false);
}

// ---------------------------------------------------------------------------
// fp32 -> bf16 elementwise convert
// ---------------------------------------------------------------------------
__global__ __launch_bounds__(256) void cvt_bf16_kernel(const float* __restrict__ x,
                                                       __bf16* __restrict__ y, int n) {
    int i = blockIdx.x * 256 + threadIdx.x;
    if (i < n) y[i] = (__bf16)x[i];
}

// Per-head weight repack: W[H=16][E=1024][D=64] fp32 -> Y[E][H*D=1024] bf16
__global__ __launch_bounds__(256) void cvt_headw_kernel(const float* __restrict__ w,
                                                        __bf16* __restrict__ y) {
    int i = blockIdx.x * 256 + threadIdx.x;   // over E*E = 1M
    int e = i >> 10;
    int hd = i & 1023;
    int h = hd >> 6;
    int d = hd & 63;
    y[i] = (__bf16)w[((size_t)h * 1024 + e) * 64 + d];
}

// ---------------------------------------------------------------------------
// WMMA GEMM: C[M,N] = A[M,K] * B[K,N]  (A,B bf16 row-major)
// Block tile 128x64, K-step 64 -> 8 WMMAs per barrier pair.
// 8 waves (256 thr), each wave 32x32 (2x2 WMMA tiles).
// Epilogue is compile-time specialized: HAS_BIAS (+ReLU), OUT_BF16.
// M%128==0, N%64==0, K%64==0 required (true for all call sites).
// ---------------------------------------------------------------------------
template <bool HAS_BIAS, bool RELU, bool OUT_BF16>
__global__ __launch_bounds__(256) void gemm_bf16_kernel(
    const __bf16* __restrict__ A, const __bf16* __restrict__ B,
    float* __restrict__ Cf, __bf16* __restrict__ Cb,
    const float* __restrict__ bias, int M, int N, int K) {
    __shared__ __align__(16) __bf16 As[128][64];
    __shared__ __align__(16) __bf16 BsT[64][64];   // B tile transposed: [n][k]

    const int t = threadIdx.x;
    const int lane = t & 31;
    const int wid = t >> 5;
    const int hi = (lane >> 4) & 1;
    const int ln = lane & 15;
    const int m0 = blockIdx.x * 128;
    const int n0 = blockIdx.y * 64;
    const int wm = (wid & 3) * 32;
    const int wn = (wid >> 2) * 32;

    v8f acc[2][2];
    for (int sm = 0; sm < 2; ++sm)
        for (int sn = 0; sn < 2; ++sn)
            for (int r = 0; r < 8; ++r) acc[sm][sn][r] = 0.0f;

    for (int kb = 0; kb < K; kb += 64) {
        // Stage A tile 128x64 (uint4 = 8 bf16 per load, 4 per thread)
#pragma unroll
        for (int i = 0; i < 4; ++i) {
            int c = t * 4 + i;            // 0..1023
            int row = c >> 3;
            int col = (c & 7) * 8;
            *(uint4*)&As[row][col] =
                *(const uint4*)(A + (size_t)(m0 + row) * K + kb + col);
        }
        // Stage B tile 64x64, transposed into BsT
#pragma unroll
        for (int i = 0; i < 16; ++i) {
            int lin = t + i * 256;        // 0..4095
            int kr = lin >> 6;
            int n = lin & 63;
            BsT[n][kr] = B[(size_t)(kb + kr) * N + n0 + n];
        }
        __syncthreads();

#pragma unroll
        for (int ks = 0; ks < 2; ++ks) {
            FragBf af[2], bf_[2];
#pragma unroll
            for (int sm = 0; sm < 2; ++sm) {
                const __bf16* ap = &As[wm + sm * 16 + ln][ks * 32];
#pragma unroll
                for (int p = 0; p < 8; ++p)
                    af[sm].u[p] = *(const unsigned int*)(ap + kmap(p, hi));
            }
#pragma unroll
            for (int sn = 0; sn < 2; ++sn) {
                const __bf16* bp = &BsT[wn + sn * 16 + ln][ks * 32];
#pragma unroll
                for (int p = 0; p < 8; ++p)
                    bf_[sn].u[p] = *(const unsigned int*)(bp + kmap(p, hi));
            }
#pragma unroll
            for (int sm = 0; sm < 2; ++sm)
#pragma unroll
                for (int sn = 0; sn < 2; ++sn)
                    acc[sm][sn] = wmma_bf16(af[sm], bf_[sn], acc[sm][sn]);
        }
        __syncthreads();
    }

    // Epilogue: C layout (ISA 7.12.2) row m = r + 8*hi (mod 16), col n = ln.
    // Bias is per-column: hoist the (at most) two values this lane needs.
    float bv[2] = {0.0f, 0.0f};
    if (HAS_BIAS) {
#pragma unroll
        for (int sn = 0; sn < 2; ++sn) bv[sn] = bias[n0 + wn + sn * 16 + ln];
    }
#pragma unroll
    for (int sm = 0; sm < 2; ++sm)
#pragma unroll
        for (int sn = 0; sn < 2; ++sn)
#pragma unroll
            for (int r = 0; r < 8; ++r) {
                int m = m0 + wm + sm * 16 + r + hi * 8;
                int n = n0 + wn + sn * 16 + ln;
                float v = acc[sm][sn][r];
                if (HAS_BIAS) v += bv[sn];
                if (RELU) v = fmaxf(v, 0.0f);
                if (OUT_BF16) Cb[(size_t)m * N + n] = (__bf16)v;
                else          Cf[(size_t)m * N + n] = v;
            }
}

// ---------------------------------------------------------------------------
// Flash attention: per block, one (b, h, 64-row q tile); 4 waves (128 thr).
// Q/K/V/O layout: [Btch*S, H*D] bf16 (head slice at column h*D).
// Online softmax in fp32 with per-row running max/sum in LDS.
// ---------------------------------------------------------------------------
__global__ __launch_bounds__(128) void flash_attn_kernel(
    const __bf16* __restrict__ Qb, const __bf16* __restrict__ Kb,
    const __bf16* __restrict__ Vb, __bf16* __restrict__ Ob,
    int Sq, int Skv, int H, int masked, float scale) {
    const int E = H * 64;
    const int q0 = blockIdx.x * 64;
    const int h = blockIdx.y;
    const int b = blockIdx.z;

    const __bf16* Qp = Qb + (size_t)b * Sq * E + h * 64;
    const __bf16* Kp = Kb + (size_t)b * Skv * E + h * 64;
    const __bf16* Vp = Vb + (size_t)b * Skv * E + h * 64;
    __bf16* Op = Ob + (size_t)b * Sq * E + h * 64;

    __shared__ __align__(16) __bf16 Ks[32][64];    // K chunk rows (k, d)
    __shared__ __align__(16) __bf16 Vt[64][32];    // V chunk transposed (d, k)
    __shared__ __align__(16) __bf16 Ps[64][32];    // P tile bf16
    __shared__ float Sc[64][33];                   // scores fp32 (padded)
    __shared__ float m_s[64], l_s[64], f_s[64];

    const int t = threadIdx.x;
    const int lane = t & 31;
    const int w = t >> 5;                          // wave id 0..3 -> q strip
    const int hi = (lane >> 4) & 1;
    const int ln = lane & 15;

    if (t < 64) { m_s[t] = -INFINITY; l_s[t] = 0.0f; }

    // Q fragments held in registers for the whole kv loop (2 k-steps over D=64)
    FragBf qf[2];
    {
        const __bf16* qp = Qp + (size_t)(q0 + w * 16 + ln) * E;
#pragma unroll
        for (int ks = 0; ks < 2; ++ks)
#pragma unroll
            for (int p = 0; p < 8; ++p)
                qf[ks].u[p] = *(const unsigned int*)(qp + ks * 32 + kmap(p, hi));
    }

    v8f o[4];
    for (int nt = 0; nt < 4; ++nt)
        for (int r = 0; r < 8; ++r) o[nt][r] = 0.0f;
    __syncthreads();

    for (int j0 = 0; j0 < Skv; j0 += 32) {
        if (masked && j0 > q0 + 63) break;         // fully-masked chunks: done

        // Stage K chunk (32x64) and V chunk transposed (64x32)
#pragma unroll
        for (int i = 0; i < 2; ++i) {
            int c = t * 2 + i;                     // 0..255 groups of 8 bf16
            int kr = c >> 3;
            int col = (c & 7) * 8;
            *(uint4*)&Ks[kr][col] = *(const uint4*)(Kp + (size_t)(j0 + kr) * E + col);
            uint4 pk = *(const uint4*)(Vp + (size_t)(j0 + kr) * E + col);
            __align__(16) __bf16 tmp8[8];
            *(uint4*)tmp8 = pk;
#pragma unroll
            for (int e = 0; e < 8; ++e) Vt[col + e][kr] = tmp8[e];
        }
        __syncthreads();

        // scores[16 x 32] = Q_strip (16x64) * K_chunk^T (64x32)
        v8f sacc[2];
        for (int nt = 0; nt < 2; ++nt)
            for (int r = 0; r < 8; ++r) sacc[nt][r] = 0.0f;
#pragma unroll
        for (int nt = 0; nt < 2; ++nt) {
            const __bf16* kp = &Ks[nt * 16 + ln][0];
#pragma unroll
            for (int ks = 0; ks < 2; ++ks) {
                FragBf bfr;
#pragma unroll
                for (int p = 0; p < 8; ++p)
                    bfr.u[p] = *(const unsigned int*)(kp + ks * 32 + kmap(p, hi));
                sacc[nt] = wmma_bf16(qf[ks], bfr, sacc[nt]);
            }
        }
#pragma unroll
        for (int nt = 0; nt < 2; ++nt)
#pragma unroll
            for (int r = 0; r < 8; ++r)
                Sc[w * 16 + r + hi * 8][nt * 16 + ln] = sacc[nt][r];
        __syncthreads();

        // Online softmax: one thread per q row (threads 0..63)
        if (t < 64) {
            float mold = m_s[t];
            int lim = masked ? ((q0 + t) - j0) : 31;   // allow jj <= lim
            if (lim > 31) lim = 31;
            float mx = -INFINITY;
            for (int jj = 0; jj <= lim; ++jj)
                mx = fmaxf(mx, Sc[t][jj] * scale);
            float mnew = fmaxf(mold, mx);
            float f = (mold > -INFINITY) ? __expf(mold - mnew) : 0.0f;
            float sum = 0.0f;
            for (int jj = 0; jj < 32; ++jj) {
                float p = (jj <= lim) ? __expf(Sc[t][jj] * scale - mnew) : 0.0f;
                Ps[t][jj] = (__bf16)p;
                sum += p;
            }
            l_s[t] = l_s[t] * f + sum;
            m_s[t] = mnew;
            f_s[t] = f;
        }
        __syncthreads();

        // O = O*f + P*V   (P: 16x32, V chunk: 32x64)
        float fr[8];
#pragma unroll
        for (int r = 0; r < 8; ++r) fr[r] = f_s[w * 16 + r + hi * 8];
        for (int nt = 0; nt < 4; ++nt)
#pragma unroll
            for (int r = 0; r < 8; ++r) o[nt][r] *= fr[r];

        FragBf pf;
        {
            const __bf16* pp = &Ps[w * 16 + ln][0];
#pragma unroll
            for (int p = 0; p < 8; ++p)
                pf.u[p] = *(const unsigned int*)(pp + kmap(p, hi));
        }
#pragma unroll
        for (int nt = 0; nt < 4; ++nt) {
            FragBf vf;
            const __bf16* vp = &Vt[nt * 16 + ln][0];
#pragma unroll
            for (int p = 0; p < 8; ++p)
                vf.u[p] = *(const unsigned int*)(vp + kmap(p, hi));
            o[nt] = wmma_bf16(pf, vf, o[nt]);
        }
        __syncthreads();
    }

    // Normalize and write out (bf16, head slice)
    float linv[8];
#pragma unroll
    for (int r = 0; r < 8; ++r) linv[r] = 1.0f / l_s[w * 16 + r + hi * 8];
    for (int nt = 0; nt < 4; ++nt)
#pragma unroll
        for (int r = 0; r < 8; ++r) {
            int rowL = w * 16 + r + hi * 8;
            int col = nt * 16 + ln;
            Op[(size_t)(q0 + rowL) * E + col] = (__bf16)(o[nt][r] * linv[r]);
        }
}

// ---------------------------------------------------------------------------
// Fused residual + LayerNorm over E=1024. pre = (x0? x0:0) + s1*x1.
// One block per row, 256 threads, 4 elems/thread. Output modes compile-time.
// ---------------------------------------------------------------------------
template <bool HAS_X0, bool OUT_F, bool OUT_B>
__global__ __launch_bounds__(256) void ln_kernel(
    const float* __restrict__ x0, const float* __restrict__ x1, float s1,
    const float* __restrict__ g, const float* __restrict__ be,
    float* __restrict__ outF, __bf16* __restrict__ outB) {
    const int E = 1024;
    const int row = blockIdx.x;
    const int t = threadIdx.x;
    const float* r1 = x1 + (size_t)row * E;
    const float* r0 = HAS_X0 ? x0 + (size_t)row * E : nullptr;

    float pre[4];
    float s = 0.0f;
#pragma unroll
    for (int i = 0; i < 4; ++i) {
        int c = t + i * 256;
        float v = s1 * r1[c] + (HAS_X0 ? r0[c] : 0.0f);
        pre[i] = v;
        s += v;
    }
    __shared__ float red[256];
    red[t] = s;
    __syncthreads();
    for (int off = 128; off > 0; off >>= 1) {
        if (t < off) red[t] += red[t + off];
        __syncthreads();
    }
    float mean = red[0] * (1.0f / E);
    __syncthreads();
    float vs = 0.0f;
#pragma unroll
    for (int i = 0; i < 4; ++i) { float d = pre[i] - mean; vs += d * d; }
    red[t] = vs;
    __syncthreads();
    for (int off = 128; off > 0; off >>= 1) {
        if (t < off) red[t] += red[t + off];
        __syncthreads();
    }
    float inv = rsqrtf(red[0] * (1.0f / E) + 1e-6f);
#pragma unroll
    for (int i = 0; i < 4; ++i) {
        int c = t + i * 256;
        float y = (pre[i] - mean) * inv * g[c] + be[c];
        if (OUT_F) outF[(size_t)row * E + c] = y;
        if (OUT_B) outB[(size_t)row * E + c] = (__bf16)y;
    }
}

// ---------------------------------------------------------------------------
// Host orchestration
// ---------------------------------------------------------------------------
extern "C" void kernel_launch(void* const* d_in, const int* in_sizes, int n_in,
                              void* d_out, int out_size, void* d_ws, size_t ws_size,
                              hipStream_t stream) {
    (void)in_sizes; (void)n_in; (void)out_size; (void)ws_size;
    const int Bc = 2, S = 2048, E = 1024, H = 16;
    const int NT = Bc * S;                 // 4096 token rows
    const int NE = NT * E;                 // 4,194,304
    const int EE = E * E;                  // 1,048,576

    const float* inputs = (const float*)d_in[0];
    const float* ctx    = (const float*)d_in[1];
    const float* Wk1 = (const float*)d_in[2];
    const float* Wv1 = (const float*)d_in[3];
    const float* Wq1 = (const float*)d_in[4];
    const float* Wo1 = (const float*)d_in[5];
    const float* bo1 = (const float*)d_in[6];
    const float* Wk2 = (const float*)d_in[7];
    const float* Wv2 = (const float*)d_in[8];
    const float* Wq2 = (const float*)d_in[9];
    const float* Wo2 = (const float*)d_in[10];
    const float* bo2 = (const float*)d_in[11];
    const float* Wf  = (const float*)d_in[12];
    const float* bf  = (const float*)d_in[13];
    const float* g1 = (const float*)d_in[14];
    const float* b1 = (const float*)d_in[15];
    const float* g2 = (const float*)d_in[16];
    const float* b2 = (const float*)d_in[17];
    const float* g3 = (const float*)d_in[18];
    const float* b3 = (const float*)d_in[19];

    char* ws = (char*)d_ws;
    size_t off = 0;
    auto take = [&](size_t bytes) -> char* {
        char* p = ws + off;
        off += (bytes + 255) & ~(size_t)255;
        return p;
    };
    __bf16* xb   = (__bf16*)take((size_t)NE * 2);
    __bf16* cxb  = (__bf16*)take((size_t)NE * 2);
    __bf16* wq1b = (__bf16*)take((size_t)EE * 2);
    __bf16* wk1b = (__bf16*)take((size_t)EE * 2);
    __bf16* wv1b = (__bf16*)take((size_t)EE * 2);
    __bf16* wo1b = (__bf16*)take((size_t)EE * 2);
    __bf16* wq2b = (__bf16*)take((size_t)EE * 2);
    __bf16* wk2b = (__bf16*)take((size_t)EE * 2);
    __bf16* wv2b = (__bf16*)take((size_t)EE * 2);
    __bf16* wo2b = (__bf16*)take((size_t)EE * 2);
    __bf16* wfb  = (__bf16*)take((size_t)EE * 2);
    __bf16* qbuf = (__bf16*)take((size_t)NE * 2);
    __bf16* kbuf = (__bf16*)take((size_t)NE * 2);
    __bf16* vbuf = (__bf16*)take((size_t)NE * 2);
    __bf16* attb = (__bf16*)take((size_t)NE * 2);
    float*  tmp  = (float*)take((size_t)NE * 4);
    __bf16* abf  = (__bf16*)take((size_t)NE * 2);
    float*  cfp  = (float*)take((size_t)NE * 4);
    __bf16* cb   = (__bf16*)take((size_t)NE * 2);

    const float scale = 1.0f / sqrtf(2048.0f);
    const dim3 gGemm(NT / 128, E / 64);
    const dim3 gAttn(S / 64, H, Bc);

    // 1) Precision conversion / weight repack
    cvt_bf16_kernel<<<NE / 256, 256, 0, stream>>>(inputs, xb, NE);
    cvt_bf16_kernel<<<NE / 256, 256, 0, stream>>>(ctx, cxb, NE);
    cvt_headw_kernel<<<EE / 256, 256, 0, stream>>>(Wq1, wq1b);
    cvt_headw_kernel<<<EE / 256, 256, 0, stream>>>(Wk1, wk1b);
    cvt_headw_kernel<<<EE / 256, 256, 0, stream>>>(Wv1, wv1b);
    cvt_headw_kernel<<<EE / 256, 256, 0, stream>>>(Wq2, wq2b);
    cvt_headw_kernel<<<EE / 256, 256, 0, stream>>>(Wk2, wk2b);
    cvt_headw_kernel<<<EE / 256, 256, 0, stream>>>(Wv2, wv2b);
    cvt_bf16_kernel<<<EE / 256, 256, 0, stream>>>(Wo1, wo1b, EE);
    cvt_bf16_kernel<<<EE / 256, 256, 0, stream>>>(Wo2, wo2b, EE);
    cvt_bf16_kernel<<<EE / 256, 256, 0, stream>>>(Wf, wfb, EE);

    // 2) Masked self-attention
    gemm_bf16_kernel<false, false, true><<<gGemm, 256, 0, stream>>>(
        xb, wq1b, nullptr, qbuf, nullptr, NT, E, E);
    gemm_bf16_kernel<false, false, true><<<gGemm, 256, 0, stream>>>(
        xb, wk1b, nullptr, kbuf, nullptr, NT, E, E);
    gemm_bf16_kernel<false, false, true><<<gGemm, 256, 0, stream>>>(
        xb, wv1b, nullptr, vbuf, nullptr, NT, E, E);
    flash_attn_kernel<<<gAttn, 128, 0, stream>>>(qbuf, kbuf, vbuf, attb, S, S, H, 1, scale);
    gemm_bf16_kernel<true, true, false><<<gGemm, 256, 0, stream>>>(
        attb, wo1b, tmp, nullptr, bo1, NT, E, E);
    ln_kernel<true, false, true><<<NT, 256, 0, stream>>>(
        inputs, tmp, 1.0f, g1, b1, nullptr, abf);

    // 3) Cross-attention (K/V from context, Q from a)
    gemm_bf16_kernel<false, false, true><<<gGemm, 256, 0, stream>>>(
        abf, wq2b, nullptr, qbuf, nullptr, NT, E, E);
    gemm_bf16_kernel<false, false, true><<<gGemm, 256, 0, stream>>>(
        cxb, wk2b, nullptr, kbuf, nullptr, NT, E, E);
    gemm_bf16_kernel<false, false, true><<<gGemm, 256, 0, stream>>>(
        cxb, wv2b, nullptr, vbuf, nullptr, NT, E, E);
    flash_attn_kernel<<<gAttn, 128, 0, stream>>>(qbuf, kbuf, vbuf, attb, S, S, H, 0, scale);
    gemm_bf16_kernel<true, true, false><<<gGemm, 256, 0, stream>>>(
        attb, wo2b, tmp, nullptr, bo2, NT, E, E);
    // faithful to source: residual adds output to itself -> pre = 2*tmp
    ln_kernel<false, true, true><<<NT, 256, 0, stream>>>(
        nullptr, tmp, 2.0f, g2, b2, cfp, cb);

    // 4) FFN + final LN
    gemm_bf16_kernel<true, true, false><<<gGemm, 256, 0, stream>>>(
        cb, wfb, tmp, nullptr, bf, NT, E, E);
    ln_kernel<true, true, false><<<NT, 256, 0, stream>>>(
        cfp, tmp, 1.0f, g3, b3, (float*)d_out, nullptr);
}
